// HeteroGraphEncoder_13125420056877
// MI455X (gfx1250) — matrix operand order repaired and checked by reading, not compiled, not run on hardware
//
#include <hip/hip_runtime.h>
#include <hip/hip_bf16.h>
#include <stdint.h>

#define H 256
#define H4 64

typedef __attribute__((ext_vector_type(16))) __bf16 v16bf;
typedef __attribute__((ext_vector_type(8)))  float  v8f;

__device__ __forceinline__ unsigned short f2bf(float f) {
  union { float f; unsigned int u; } c; c.f = f;
  unsigned int u = c.u;
  unsigned int r = u + 0x7FFFu + ((u >> 16) & 1u);   // round-to-nearest-even
  return (unsigned short)(r >> 16);
}

// ---------------------------------------------------------------------------
// Weight transpose + fp32 -> bf16: Wt[n*K + k] = bf16(W[k*N + n]); N = 256
// ---------------------------------------------------------------------------
__global__ void cvtw_kernel(const float* __restrict__ W, unsigned short* __restrict__ Wt,
                            int K, int N) {
  int i = blockIdx.x * blockDim.x + threadIdx.x;
  if (i >= N * K) return;
  int n = i / K, k = i - n * K;
  Wt[i] = f2bf(W[(size_t)k * N + n]);
}

__global__ void fill_kernel(float* __restrict__ p, long long n) {
  long long i = (long long)blockIdx.x * blockDim.x + threadIdx.x;
  if (i < n) p[i] = 0.0f;
}

// ---------------------------------------------------------------------------
// Activation convert (memory-bound): out_bf16 = bf16(x (+ alpha*y)), 4 elems/thread
// ---------------------------------------------------------------------------
__global__ void cvt_act_kernel(const float* __restrict__ x, const float* __restrict__ y,
                               float alpha, unsigned short* __restrict__ out,
                               long long n4) {
  long long i = (long long)blockIdx.x * blockDim.x + threadIdx.x;
  if (i >= n4) return;
  float4 v = ((const float4*)x)[i];
  if (y) {
    float4 w = ((const float4*)y)[i];
    v.x += alpha * w.x; v.y += alpha * w.y; v.z += alpha * w.z; v.w += alpha * w.w;
  }
  ushort4 o;
  o.x = f2bf(v.x); o.y = f2bf(v.y); o.z = f2bf(v.z); o.w = f2bf(v.w);
  ((ushort4*)out)[i] = o;
}

// ---------------------------------------------------------------------------
// WMMA GEMM: Out[M,N] = relu(A_bf16[M,KT] @ Wt^T + bias), KT compile-time (64/256).
// Wt is bf16 [N][KT] (pre-transposed). Block = 256 threads = 8 waves.
// Waves 0-3: rows m0..m0+15, cols (w&3)*64..+63 ; waves 4-7: rows +16.
// K-loop fully unrolled: all addresses are static offsets, letting the scheduler
// keep multiple B fragments live and overlap next-step loads with WMMAs.
// ---------------------------------------------------------------------------
template <int KT, bool OUT_BF16>
__global__ __launch_bounds__(256) void gemm_wmma_kernel(
    const unsigned short* __restrict__ A, const unsigned short* __restrict__ Wt,
    const float* __restrict__ bias, void* __restrict__ Out, int M, int N) {
  const int lane = threadIdx.x & 31;
  const int wave = threadIdx.x >> 5;
  const int half = lane >> 4;       // 0: lanes 0-15, 1: lanes 16-31
  const int lrow = lane & 15;
  const int m0 = blockIdx.x * 32 + ((wave >> 2) << 4);
  const int n0 = (wave & 3) * 64;
  const int m  = m0 + lrow;
  const int mc = (m < M) ? m : (M - 1);   // clamp OOB rows; stores are guarded

  v8f acc[4] = {};

  union AU { uint4 q[2]; v16bf b; };

  // ISA 16-bit A layout: lane half 0 -> K 0..7 / 16..23 ; half 1 -> 8..15 / 24..31
  const unsigned short* Ar  = A  + (size_t)mc * KT + 8 * half;
  // ISA 16-bit B layout: lane half 0 -> K 0..7 / 16..23 ; half 1 -> 8..15 / 24..31
  const unsigned short* Br0 = Wt + (size_t)(n0 + lrow) * KT + 8 * half;

#pragma unroll
  for (int kb = 0; kb < KT; kb += 32) {
    AU a;
    a.q[0] = *(const uint4*)(Ar + kb);        // K = kb + 8*half + 0..7
    a.q[1] = *(const uint4*)(Ar + kb + 16);   // K = kb + 16 + 8*half + 0..7

    AU bf[4];
#pragma unroll
    for (int nt = 0; nt < 4; ++nt) {
      const unsigned short* Br = Br0 + nt * (16 * KT) + kb;
      bf[nt].q[0] = *(const uint4*)(Br);      // K = kb + 8*half + 0..7
      bf[nt].q[1] = *(const uint4*)(Br + 16); // K = kb + 16 + 8*half + 0..7
    }
#pragma unroll
    for (int nt = 0; nt < 4; ++nt)
      acc[nt] = __builtin_amdgcn_wmma_f32_16x16x32_bf16(
          false, a.b, false, bf[nt].b, (short)0, acc[nt], false, false);
  }

#pragma unroll
  for (int nt = 0; nt < 4; ++nt) {
    const int col = n0 + nt * 16 + lrow;
    const float bv = bias[col];
#pragma unroll
    for (int r = 0; r < 8; ++r) {
      const int row = m0 + (half << 3) + r;   // C layout: lanes 16-31 -> M = r+8
      if (row < M) {
        float v = fmaxf(acc[nt][r] + bv, 0.0f);
        if constexpr (OUT_BF16)
          ((unsigned short*)Out)[(size_t)row * N + col] = f2bf(v);
        else
          ((float*)Out)[(size_t)row * N + col] = v;
      }
    }
  }
}

// ---------------------------------------------------------------------------
// Graph conv machinery (memory-bound; float4 gathers + f32 atomics)
// out[tgt[e]] += emb[src[e]]  over E edges x 256 floats
// ---------------------------------------------------------------------------
__global__ void scatter_add_kernel(const float* __restrict__ emb,
                                   const int* __restrict__ sidx,
                                   const int* __restrict__ tidx,
                                   float* __restrict__ out, int E) {
  long long i = (long long)blockIdx.x * blockDim.x + threadIdx.x;
  if (i >= (long long)E * H4) return;
  int e = (int)(i >> 6);
  int c = (int)(i & 63);
  int s = sidx[e], t = tidx[e];
  const float4* src = (const float4*)(emb + (size_t)s * H) + c;
  __builtin_prefetch(src, 0, 1);            // global_prefetch_b8 (speculative)
  float4 v = *src;
  float* o = out + (size_t)t * H + c * 4;
  atomicAdd(o + 0, v.x);
  atomicAdd(o + 1, v.y);
  atomicAdd(o + 2, v.z);
  atomicAdd(o + 3, v.w);
}

__global__ void count_kernel(const int* __restrict__ tidx, float* __restrict__ cnt, int E) {
  int e = blockIdx.x * blockDim.x + threadIdx.x;
  if (e < E) atomicAdd(&cnt[tidx[e]], 1.0f);
}

// x += alpha * m / max(cnt, 1)
__global__ void apply_mean_kernel(float* __restrict__ x, const float* __restrict__ m,
                                  const float* __restrict__ cnt, float alpha, int rows) {
  long long i = (long long)blockIdx.x * blockDim.x + threadIdx.x;
  if (i >= (long long)rows * H4) return;
  int r = (int)(i >> 6);
  int c = (int)(i & 63);
  float s = alpha / fmaxf(cnt[r], 1.0f);
  float4* xp = (float4*)(x + (size_t)r * H) + c;
  const float4* mp = (const float4*)(m + (size_t)r * H) + c;
  float4 xv = *xp;
  float4 mv = *mp;
  xv.x += s * mv.x; xv.y += s * mv.y; xv.z += s * mv.z; xv.w += s * mv.w;
  *xp = xv;
}

// Partial column sums: block handles 64 rows, thread j owns column j (coalesced)
__global__ __launch_bounds__(256) void colsum_kernel(const float* __restrict__ x,
                                                     float* __restrict__ acc, int M) {
  int j = threadIdx.x;
  long long r0 = (long long)blockIdx.x * 64;
  long long rend = r0 + 64; if (rend > M) rend = M;
  float s = 0.0f;
  for (long long r = r0; r < rend; ++r) s += x[r * H + j];
  atomicAdd(&acc[j], s);
}

// ---------------------------------------------------------------------------
// Head MLP (tiny): means -> two parallel 256x256 GEMVs -> 512x256 -> 256x256
// ---------------------------------------------------------------------------
__global__ __launch_bounds__(256) void head_kernel(
    const float* __restrict__ acc_t, const float* __restrict__ acc_e,
    float invNT, float invNE,
    const float* __restrict__ ta_w, const float* __restrict__ ta_b,
    const float* __restrict__ ea_w, const float* __restrict__ ea_b,
    const float* __restrict__ ow1, const float* __restrict__ ob1,
    const float* __restrict__ ow2, const float* __restrict__ ob2,
    float* __restrict__ out) {
  __shared__ float mt[H], me[H], comb[2 * H], hid[H];
  int j = threadIdx.x;
  mt[j] = acc_t[j] * invNT;
  me[j] = acc_e[j] * invNE;
  __syncthreads();
  float s1 = ta_b[j], s2 = ea_b[j];
  for (int k = 0; k < H; ++k) {
    s1 += mt[k] * ta_w[k * H + j];
    s2 += me[k] * ea_w[k * H + j];
  }
  comb[j]     = fmaxf(s1, 0.0f);
  comb[H + j] = fmaxf(s2, 0.0f);
  __syncthreads();
  float h1 = ob1[j];
  for (int k = 0; k < 2 * H; ++k) h1 += comb[k] * ow1[k * H + j];
  hid[j] = fmaxf(h1, 0.0f);
  __syncthreads();
  float o = ob2[j];
  for (int k = 0; k < H; ++k) o += hid[k] * ow2[k * H + j];
  out[j] = o;
}

// ---------------------------------------------------------------------------
extern "C" void kernel_launch(void* const* d_in, const int* in_sizes, int n_in,
                              void* d_out, int out_size, void* d_ws, size_t ws_size,
                              hipStream_t stream) {
  (void)n_in; (void)out_size; (void)ws_size;
  const float* x_t   = (const float*)d_in[0];
  const float* x_e   = (const float*)d_in[1];
  const int*   qe    = (const int*)d_in[2];
  const int*   ty    = (const int*)d_in[3];
  const int*   af    = (const int*)d_in[4];
  const int*   tp    = (const int*)d_in[5];
  const float* te_w1 = (const float*)d_in[6];  const float* te_b1 = (const float*)d_in[7];
  const float* te_w2 = (const float*)d_in[8];  const float* te_b2 = (const float*)d_in[9];
  const float* ee_w1 = (const float*)d_in[10]; const float* ee_b1 = (const float*)d_in[11];
  const float* ee_w2 = (const float*)d_in[12]; const float* ee_b2 = (const float*)d_in[13];
  const float* g_w0  = (const float*)d_in[14]; const float* g_b0  = (const float*)d_in[15];
  const float* g_w1  = (const float*)d_in[16]; const float* g_b1  = (const float*)d_in[17];
  const float* ta_w  = (const float*)d_in[18]; const float* ta_b  = (const float*)d_in[19];
  const float* ea_w  = (const float*)d_in[20]; const float* ea_b  = (const float*)d_in[21];
  const float* ow1   = (const float*)d_in[22]; const float* ob1   = (const float*)d_in[23];
  const float* ow2   = (const float*)d_in[24]; const float* ob2   = (const float*)d_in[25];

  const int NT = in_sizes[0] / 64;
  const int NE = in_sizes[1] / 64;
  const int E  = in_sizes[2] / 2;
  const int NMAX = NT > NE ? NT : NE;

  char* p = (char*)d_ws;
  auto carve = [&](size_t b) -> char* {
    char* r = p; p += (b + 255) & ~(size_t)255; return r;
  };
  unsigned short* wte1 = (unsigned short*)carve((size_t)256 * 64 * 2);
  unsigned short* wte2 = (unsigned short*)carve((size_t)256 * 256 * 2);
  unsigned short* wee1 = (unsigned short*)carve((size_t)256 * 64 * 2);
  unsigned short* wee2 = (unsigned short*)carve((size_t)256 * 256 * 2);
  unsigned short* wg0  = (unsigned short*)carve((size_t)256 * 256 * 2);
  unsigned short* wg1  = (unsigned short*)carve((size_t)256 * 256 * 2);
  float* tbuf = (float*)carve((size_t)NT * H * 4);
  float* ebuf = (float*)carve((size_t)NE * H * 4);
  float* qmsg = (float*)carve((size_t)NMAX * H * 4);   // also affinity e-msgs
  float* mbuf = (float*)carve((size_t)NMAX * H * 4);
  unsigned short* abuf = (unsigned short*)carve((size_t)NMAX * H * 2); // bf16 GEMM A staging
  char*  hsh  = carve((size_t)NMAX * H * 4);           // h1(bf16) then GNN hidden(f32)
  unsigned short* h1bf = (unsigned short*)hsh;
  float*          hbuf = (float*)hsh;
  float* cnt1  = (float*)carve((size_t)NMAX * 4);
  float* cnt2  = (float*)carve((size_t)NMAX * 4);
  float* acc_t = (float*)carve((size_t)H * 4);
  float* acc_e = (float*)carve((size_t)H * 4);

  const int TB = 256;
  auto nb = [&](long long n) { return (unsigned)((n + TB - 1) / TB); };

  // --- weights -> transposed bf16
  cvtw_kernel<<<nb(256 * 64),  TB, 0, stream>>>(te_w1, wte1, 64, 256);
  cvtw_kernel<<<nb(256 * 256), TB, 0, stream>>>(te_w2, wte2, 256, 256);
  cvtw_kernel<<<nb(256 * 64),  TB, 0, stream>>>(ee_w1, wee1, 64, 256);
  cvtw_kernel<<<nb(256 * 256), TB, 0, stream>>>(ee_w2, wee2, 256, 256);
  cvtw_kernel<<<nb(256 * 256), TB, 0, stream>>>(g_w0,  wg0,  256, 256);
  cvtw_kernel<<<nb(256 * 256), TB, 0, stream>>>(g_w1,  wg1,  256, 256);

  dim3 gT((NT + 31) / 32), gE((NE + 31) / 32);
  // --- task encoder
  cvt_act_kernel<<<nb((long long)NT * 16), TB, 0, stream>>>(x_t, nullptr, 0.f, abuf, (long long)NT * 16);
  gemm_wmma_kernel<64,  true ><<<gT, TB, 0, stream>>>(abuf, wte1, te_b1, h1bf, NT, 256);
  gemm_wmma_kernel<256, false><<<gT, TB, 0, stream>>>(h1bf, wte2, te_b2, tbuf, NT, 256);
  // --- edge encoder
  cvt_act_kernel<<<nb((long long)NE * 16), TB, 0, stream>>>(x_e, nullptr, 0.f, abuf, (long long)NE * 16);
  gemm_wmma_kernel<64,  true ><<<gE, TB, 0, stream>>>(abuf, wee1, ee_b1, h1bf, NE, 256);
  gemm_wmma_kernel<256, false><<<gE, TB, 0, stream>>>(h1bf, wee2, ee_b2, ebuf, NE, 256);

  // --- queue conv: q_msgs from original t, then 2 GNN layers (add fused in cvt)
  fill_kernel<<<nb((long long)NT * H), TB, 0, stream>>>(qmsg, (long long)NT * H);
  scatter_add_kernel<<<nb((long long)E * H4), TB, 0, stream>>>(tbuf, qe, qe + E, qmsg, E);
  cvt_act_kernel<<<nb((long long)NT * H4), TB, 0, stream>>>(tbuf, qmsg, 0.5f, abuf, (long long)NT * H4);
  gemm_wmma_kernel<256, false><<<gT, TB, 0, stream>>>(abuf, wg0, g_b0, hbuf, NT, 256);
  cvt_act_kernel<<<nb((long long)NT * H4), TB, 0, stream>>>(hbuf, qmsg, 0.5f, abuf, (long long)NT * H4);
  gemm_wmma_kernel<256, false><<<gT, TB, 0, stream>>>(abuf, wg1, g_b1, tbuf, NT, 256);

  // --- type conv (mean)
  fill_kernel<<<nb((long long)NT * H), TB, 0, stream>>>(mbuf, (long long)NT * H);
  fill_kernel<<<nb(NT), TB, 0, stream>>>(cnt1, NT);
  scatter_add_kernel<<<nb((long long)E * H4), TB, 0, stream>>>(tbuf, ty, ty + E, mbuf, E);
  count_kernel<<<nb(E), TB, 0, stream>>>(ty + E, cnt1, E);
  apply_mean_kernel<<<nb((long long)NT * H4), TB, 0, stream>>>(tbuf, mbuf, cnt1, 0.3f, NT);

  // --- affinity conv (simultaneous bipartite update: both msg sets first)
  fill_kernel<<<nb((long long)NT * H), TB, 0, stream>>>(mbuf, (long long)NT * H);
  fill_kernel<<<nb((long long)NE * H), TB, 0, stream>>>(qmsg, (long long)NE * H);
  fill_kernel<<<nb(NT), TB, 0, stream>>>(cnt1, NT);
  fill_kernel<<<nb(NE), TB, 0, stream>>>(cnt2, NE);
  scatter_add_kernel<<<nb((long long)E * H4), TB, 0, stream>>>(ebuf, af + E, af, mbuf, E); // e[atgt] -> asrc
  count_kernel<<<nb(E), TB, 0, stream>>>(af, cnt1, E);
  scatter_add_kernel<<<nb((long long)E * H4), TB, 0, stream>>>(tbuf, af, af + E, qmsg, E); // t[asrc] -> atgt
  count_kernel<<<nb(E), TB, 0, stream>>>(af + E, cnt2, E);
  apply_mean_kernel<<<nb((long long)NT * H4), TB, 0, stream>>>(tbuf, mbuf, cnt1, 0.3f, NT);
  apply_mean_kernel<<<nb((long long)NE * H4), TB, 0, stream>>>(ebuf, qmsg, cnt2, 0.3f, NE);

  // --- topology conv (mean)
  fill_kernel<<<nb((long long)NE * H), TB, 0, stream>>>(mbuf, (long long)NE * H);
  fill_kernel<<<nb(NE), TB, 0, stream>>>(cnt1, NE);
  scatter_add_kernel<<<nb((long long)E * H4), TB, 0, stream>>>(ebuf, tp, tp + E, mbuf, E);
  count_kernel<<<nb(E), TB, 0, stream>>>(tp + E, cnt1, E);
  apply_mean_kernel<<<nb((long long)NE * H4), TB, 0, stream>>>(ebuf, mbuf, cnt1, 0.3f, NE);

  // --- aggregation + head
  fill_kernel<<<1, TB, 0, stream>>>(acc_t, H);
  fill_kernel<<<1, TB, 0, stream>>>(acc_e, H);
  colsum_kernel<<<(NT + 63) / 64, TB, 0, stream>>>(tbuf, acc_t, NT);
  colsum_kernel<<<(NE + 63) / 64, TB, 0, stream>>>(ebuf, acc_e, NE);
  head_kernel<<<1, TB, 0, stream>>>(acc_t, acc_e, 1.0f / NT, 1.0f / NE,
                                    ta_w, ta_b, ea_w, ea_b, ow1, ob1, ow2, ob2,
                                    (float*)d_out);
}
